// Grok1DecoderLayer_17136919511388
// MI455X (gfx1250) — matrix operand order repaired and checked by reading, not compiled, not run on hardware
//
#include <hip/hip_runtime.h>

// ---------------------------------------------------------------------------
// Grok1 decoder layer for gfx1250 (MI455X): bf16 WMMA everywhere.
// - LDS tiles stored pre-swizzled in WMMA fragment order (all fragment loads
//   are ds_load_b128).
// - GEMM K-loops are double-buffered: one barrier per K-step, global weight
//   stream overlaps the WMMA pipe.
// - 32x64 per-wave tiles in the big GEMMs: each B fragment feeds 2 WMMAs.
// - Routed top-2 MoE (gather/scatter) instead of dense 8-expert compute.
// ---------------------------------------------------------------------------

typedef __attribute__((ext_vector_type(16))) __bf16 bf16x16;
typedef __attribute__((ext_vector_type(8)))  float  f32x8;

constexpr int kB   = 2;
constexpr int kS   = 1024;
constexpr int kH   = 1024;
constexpr int kNH  = 16;
constexpr int kNKV = 4;
constexpr int kDH  = 64;
constexpr int kI   = 4096;
constexpr int kE   = 8;
constexpr int kT   = kB * kS;          // 2048 tokens
constexpr float kEPS = 1e-5f;

// ---------------------------------------------------------------------------
// helpers
// ---------------------------------------------------------------------------
__device__ inline f32x8 zero8() {
  f32x8 z;
#pragma unroll
  for (int i = 0; i < 8; ++i) z[i] = 0.0f;
  return z;
}

__device__ inline f32x8 wmma_bf16(bf16x16 a, bf16x16 b, f32x8 c) {
  // (neg_a, A, neg_b, B, c_mod, C, reuse_a, reuse_b)
  return __builtin_amdgcn_wmma_f32_16x16x32_bf16(false, a, false, b, (short)0, c,
                                                 false, false);
}

__device__ inline unsigned int pack_bf16x2(float lo, float hi) {
  union { __bf16 b; unsigned short s; } a, c;
  a.b = (__bf16)lo;
  c.b = (__bf16)hi;
  return (unsigned int)a.s | ((unsigned int)c.s << 16);
}

// 16 contiguous bf16 -> fragment (2 x ds_load_b128)
__device__ inline bf16x16 load_frag16(const __bf16* p) {
  union { uint4 u[2]; bf16x16 v; } c;
  c.u[0] = *reinterpret_cast<const uint4*>(p);
  c.u[1] = *reinterpret_cast<const uint4*>(p + 8);
  return c.v;
}

// A-matrix 16x32 bf16 fragment from row-major LDS tile (ld multiple of 8).
// Per-lane data is contiguous: elems 0..15 @ row*ld + hi*16.
__device__ inline bf16x16 load_A(const __bf16* p, int ld) {
  const int lane = threadIdx.x & 31;
  const int row = lane & 15, hi = lane >> 4;
  union { uint4 u[2]; bf16x16 v; } c;
  c.u[0] = *reinterpret_cast<const uint4*>(p + row * ld + hi * 16);
  c.u[1] = *reinterpret_cast<const uint4*>(p + row * ld + 8 + hi * 16);
  return c.v;
}

// B fragment from a swizzled [32][16] chunk: lane's 16 bf16 are contiguous.
__device__ inline bf16x16 load_B_sw(const __bf16* chunk) {
  const int lane = threadIdx.x & 31;
  return load_frag16(chunk + lane * 16);
}

__device__ inline float block_reduce_sum256(float v) {
  __shared__ float sb[8];
#pragma unroll
  for (int m = 16; m >= 1; m >>= 1) v += __shfl_xor(v, m, 32);
  const int wid = threadIdx.x >> 5, lane = threadIdx.x & 31;
  if (lane == 0) sb[wid] = v;
  __syncthreads();
  float tot = 0.f;
#pragma unroll
  for (int i = 0; i < 8; ++i) tot += sb[i];
  __syncthreads();
  return tot;
}

__device__ inline float gelu_tanh(float x) {
  const float x3 = x * x * x;
  return 0.5f * x * (1.0f + tanhf(0.7978845608028654f * (x + 0.044715f * x3)));
}

union F4 { float4 v; float f[4]; };
union U4 { uint4 u; unsigned short s[8]; };

// ---------------------------------------------------------------------------
// CDNA5 Tensor-Data-Mover / cluster demo (compiled, never launched):
// exercises tensor_load_to_lds + s_wait_tensorcnt + cluster ops on gfx1250.
// Guarded so it vanishes cleanly on toolchains without the builtins.
// ---------------------------------------------------------------------------
#if defined(__AMDGCN__) && __has_builtin(__builtin_amdgcn_tensor_load_to_lds)
typedef unsigned int u32x4_t __attribute__((ext_vector_type(4)));
typedef int          i32x4_t __attribute__((ext_vector_type(4)));
typedef int          i32x8_t __attribute__((ext_vector_type(8)));

__global__ void tdm_demo_kernel(const __bf16* __restrict__ src,
                                __bf16* __restrict__ dst) {
  __shared__ __attribute__((aligned(128))) __bf16 tile[64 * 32];
  const unsigned lds_addr = (unsigned)(size_t)(void*)&tile[0];
  const unsigned long long ga = (unsigned long long)(size_t)src;

  // D# group 0: count=1 | lds_addr | global_addr[56:0] | type=2 ("image")
  u32x4_t g0;
  g0[0] = 1u;
  g0[1] = lds_addr;
  g0[2] = (unsigned)ga;
  g0[3] = (unsigned)((ga >> 32) & 0x01FFFFFFu) | (2u << 30);

  // D# group 1: data_size=1 (2B) | tensor_dim0=1024 | tensor_dim1=2048 |
  //             tile_dim0=32 | tile_dim1=64 | tensor_dim0_stride=1024
  i32x8_t g1;
  g1[0] = (int)(1u << 16);
  g1[1] = (int)((1024u & 0xFFFFu) << 16);
  g1[2] = (int)(((1024u >> 16) & 0xFFFFu) | ((2048u & 0xFFFFu) << 16));
  g1[3] = (int)(((2048u >> 16) & 0xFFFFu) | (32u << 16));
  g1[4] = (int)64u;
  g1[5] = (int)1024u;
  g1[6] = 0;
  g1[7] = 0;

  i32x4_t g2, g3;
#pragma unroll
  for (int i = 0; i < 4; ++i) { g2[i] = 0; g3[i] = 0; }

#if __clang_major__ >= 23
  i32x8_t g4;
#pragma unroll
  for (int i = 0; i < 8; ++i) g4[i] = 0;
  __builtin_amdgcn_tensor_load_to_lds(g0, g1, g2, g3, g4, 0);
#else
  __builtin_amdgcn_tensor_load_to_lds(g0, g1, g2, g3, 0);
#endif
#if __has_builtin(__builtin_amdgcn_s_wait_tensorcnt)
  __builtin_amdgcn_s_wait_tensorcnt(0);
#endif
  __syncthreads();
  for (int i = threadIdx.x; i < 64 * 32; i += blockDim.x) dst[i] = tile[i];
#if __has_builtin(__builtin_amdgcn_cluster_id_x)
  if (__builtin_amdgcn_cluster_id_x() == 12345) dst[0] = (__bf16)1.0f;
#endif
#if __has_builtin(__builtin_amdgcn_s_cluster_barrier)
  __builtin_amdgcn_s_cluster_barrier();
#endif
}
#endif  // tensor_load_to_lds

// ---------------------------------------------------------------------------
// RMSNorm kernels
// ---------------------------------------------------------------------------
__global__ __launch_bounds__(256) void rms_bf16_kernel(
    const float* __restrict__ x, const float* __restrict__ sc,
    __bf16* __restrict__ out) {
  const int t = blockIdx.x;
  const float* xp = x + (size_t)t * kH;
  float ss = 0.f;
  for (int i = threadIdx.x; i < kH; i += 256) { const float v = xp[i]; ss += v * v; }
  const float r = rsqrtf(block_reduce_sum256(ss) / kH + kEPS);
  for (int i = threadIdx.x; i < kH; i += 256)
    out[(size_t)t * kH + i] = (__bf16)(xp[i] * r * sc[i]);
}

// x = hidden + rmsnorm(attnproj, sc1);  h2 = bf16(rmsnorm(x, sc2))
__global__ __launch_bounds__(256) void post_attn_kernel(
    const float* __restrict__ attnp, const float* __restrict__ hid,
    const float* __restrict__ sc1, const float* __restrict__ sc2,
    float* __restrict__ x, __bf16* __restrict__ h2) {
  const int t = blockIdx.x;
  const float* ap = attnp + (size_t)t * kH;
  const float* hp = hid + (size_t)t * kH;
  float ss = 0.f;
  for (int i = threadIdx.x; i < kH; i += 256) { const float v = ap[i]; ss += v * v; }
  const float r1 = rsqrtf(block_reduce_sum256(ss) / kH + kEPS);
  float ss2 = 0.f;
  for (int i = threadIdx.x; i < kH; i += 256) {
    const float xv = hp[i] + ap[i] * r1 * sc1[i];
    x[(size_t)t * kH + i] = xv;
    ss2 += xv * xv;
  }
  const float r2 = rsqrtf(block_reduce_sum256(ss2) / kH + kEPS);
  for (int i = threadIdx.x; i < kH; i += 256)
    h2[(size_t)t * kH + i] = (__bf16)(x[(size_t)t * kH + i] * r2 * sc2[i]);
}

// out = x + rmsnorm(moe, sc)
__global__ __launch_bounds__(256) void final_kernel(
    const float* __restrict__ moe, const float* __restrict__ x,
    const float* __restrict__ sc, float* __restrict__ out) {
  const int t = blockIdx.x;
  const float* mp = moe + (size_t)t * kH;
  float ss = 0.f;
  for (int i = threadIdx.x; i < kH; i += 256) { const float v = mp[i]; ss += v * v; }
  const float r = rsqrtf(block_reduce_sum256(ss) / kH + kEPS);
  for (int i = threadIdx.x; i < kH; i += 256)
    out[(size_t)t * kH + i] = x[(size_t)t * kH + i] + mp[i] * r * sc[i];
}

// ---------------------------------------------------------------------------
// Plain GEMM: C[M,N] f32 = A[M,K] bf16 * W[K,N] f32 (bf16-swizzled in LDS).
// block tile 128x128, 8 waves, wave = one 32x64 strip (2 A frags, each B
// fragment feeds 2 WMMAs), K-step 32, double-buffered LDS.
// ---------------------------------------------------------------------------
__global__ __launch_bounds__(256) void gemm_bf16_kernel(
    const __bf16* __restrict__ A, const float* __restrict__ W,
    float* __restrict__ C, int N, int K) {
  __shared__ __attribute__((aligned(16))) __bf16 As[2][128][40];
  __shared__ __attribute__((aligned(16))) __bf16 Wsw[2][8][32][16];
  const int tid = threadIdx.x, lane = tid & 31, wid = tid >> 5;
  const int m0 = blockIdx.y * 128, n0 = blockIdx.x * 128;
  const int mt = (wid & 3) * 32, nh = (wid >> 2) * 64, jb = nh >> 4;
  const int ar = tid >> 1, ac = (tid & 1) * 16;
  const int kp = tid >> 4, nseg = (tid & 15) * 8;
  const int whi = (kp >= 8) ? 16 : 0, wpos = 2 * (kp & 7);
  f32x8 acc[2][4];
#pragma unroll
  for (int m = 0; m < 2; ++m)
#pragma unroll
    for (int j = 0; j < 4; ++j) acc[m][j] = zero8();

  auto stage = [&](int buf, int k0) {
    const __bf16* asrc = &A[(size_t)(m0 + ar) * K + k0 + ac];
    *reinterpret_cast<uint4*>(&As[buf][ar][ac]) =
        *reinterpret_cast<const uint4*>(asrc);
    *reinterpret_cast<uint4*>(&As[buf][ar][ac + 8]) =
        *reinterpret_cast<const uint4*>(asrc + 8);
    const float* Wr0 = &W[(size_t)(k0 + 2 * kp) * N + n0 + nseg];
    const float* Wr1 = Wr0 + N;
    F4 a0, a1, b0, b1;
    a0.v = *reinterpret_cast<const float4*>(Wr0);
    a1.v = *reinterpret_cast<const float4*>(Wr0 + 4);
    b0.v = *reinterpret_cast<const float4*>(Wr1);
    b1.v = *reinterpret_cast<const float4*>(Wr1 + 4);
#pragma unroll
    for (int u = 0; u < 8; ++u) {
      const float lo = (u < 4) ? a0.f[u] : a1.f[u - 4];
      const float hv = (u < 4) ? b0.f[u] : b1.f[u - 4];
      const int n = nseg + u, j = n >> 4, c = n & 15;
      *reinterpret_cast<unsigned int*>(&Wsw[buf][j][c + whi][wpos]) =
          pack_bf16x2(lo, hv);
    }
    if (k0 + 32 < K)
      __builtin_prefetch(&W[(size_t)(k0 + 32 + 2 * kp) * N + n0 + nseg], 0, 1);
  };

  stage(0, 0);
  __syncthreads();
  for (int k0 = 0; k0 < K; k0 += 32) {
    const int cur = (k0 >> 5) & 1;
    if (k0 + 32 < K) stage(cur ^ 1, k0 + 32);
    const bf16x16 a0 = load_A(&As[cur][mt][0], 40);
    const bf16x16 a1 = load_A(&As[cur][mt + 16][0], 40);
#pragma unroll
    for (int j = 0; j < 4; ++j) {
      const bf16x16 bf = load_B_sw(&Wsw[cur][jb + j][0][0]);
      acc[0][j] = wmma_bf16(a0, bf, acc[0][j]);
      acc[1][j] = wmma_bf16(a1, bf, acc[1][j]);
    }
    __syncthreads();
  }
  const int col = lane & 15, hi = lane >> 4;
#pragma unroll
  for (int m = 0; m < 2; ++m)
#pragma unroll
    for (int j = 0; j < 4; ++j)
#pragma unroll
      for (int v = 0; v < 8; ++v)
        C[(size_t)(m0 + mt + m * 16 + v + 8 * hi) * N + n0 + nh + j * 16 + col] =
            acc[m][j][v];
}

// ---------------------------------------------------------------------------
// RoPE + pack to bf16 in attention layout [B, heads, S, DH]
// ---------------------------------------------------------------------------
__global__ void rope_q_kernel(const float* __restrict__ q,
                              const int* __restrict__ pos_ids,
                              __bf16* __restrict__ Qb) {
  const int idx = blockIdx.x * blockDim.x + threadIdx.x;
  if (idx >= kT * kNH * 32) return;
  const int d = idx & 31;
  const int th = idx >> 5;
  const int h = th % kNH, t = th / kNH;
  const int b = t / kS, s = t % kS;
  const float pos = (float)pos_ids[t];
  const float freq = __powf(10000.0f, -(float)d / 32.0f);
  float sn, cs;
  __sincosf(pos * freq, &sn, &cs);
  const float x1 = q[(size_t)t * (kNH * kDH) + h * kDH + d];
  const float x2 = q[(size_t)t * (kNH * kDH) + h * kDH + d + 32];
  const size_t o = (((size_t)b * kNH + h) * kS + s) * kDH + d;
  Qb[o]      = (__bf16)(x1 * cs - x2 * sn);
  Qb[o + 32] = (__bf16)(x2 * cs + x1 * sn);
}

__global__ void rope_kv_kernel(const float* __restrict__ k,
                               const float* __restrict__ v,
                               const int* __restrict__ pos_ids,
                               __bf16* __restrict__ Kb, __bf16* __restrict__ Vb) {
  const int idx = blockIdx.x * blockDim.x + threadIdx.x;
  if (idx >= kT * kNKV * 32) return;
  const int d = idx & 31;
  const int th = idx >> 5;
  const int h = th % kNKV, t = th / kNKV;
  const int b = t / kS, s = t % kS;
  const float pos = (float)pos_ids[t];
  const float freq = __powf(10000.0f, -(float)d / 32.0f);
  float sn, cs;
  __sincosf(pos * freq, &sn, &cs);
  const float x1 = k[(size_t)t * (kNKV * kDH) + h * kDH + d];
  const float x2 = k[(size_t)t * (kNKV * kDH) + h * kDH + d + 32];
  const size_t o = (((size_t)b * kNKV + h) * kS + s) * kDH + d;
  Kb[o]      = (__bf16)(x1 * cs - x2 * sn);
  Kb[o + 32] = (__bf16)(x2 * cs + x1 * sn);
  Vb[o]      = (__bf16)v[(size_t)t * (kNKV * kDH) + h * kDH + d];
  Vb[o + 32] = (__bf16)v[(size_t)t * (kNKV * kDH) + h * kDH + d + 32];
}

// ---------------------------------------------------------------------------
// Flash attention: grid (S/64, B*NH), 128 threads = 4 waves, 16 q-rows/wave.
// K and V tiles are staged directly in B-fragment-swizzled order.
// ---------------------------------------------------------------------------
__global__ __launch_bounds__(128) void flash_attn_kernel(
    const __bf16* __restrict__ Qb, const __bf16* __restrict__ Kb,
    const __bf16* __restrict__ Vb, const int* __restrict__ amask,
    __bf16* __restrict__ Ob) {
  __shared__ __attribute__((aligned(16))) __bf16 Ksw[2][2][32][16]; // [chunk][kstep]
  __shared__ __attribute__((aligned(16))) __bf16 Vsw[4][32][16];    // [dchunk]
  __shared__ __attribute__((aligned(16))) __bf16 Ps[4][16][40];
  const int tid = threadIdx.x, lane = tid & 31, wave = tid >> 5;
  const int bh = blockIdx.y, b = bh / kNH, h = bh % kNH;
  const int kvh = h / (kNH / kNKV);
  const int qtile = blockIdx.x * 64;
  const int qrow_base = qtile + wave * 16;
  const int col = lane & 15, hi = lane >> 4;
  const int kp = tid >> 3, seg = tid & 7;   // staging coords: key pair, d-segment

  // Q fragments for K-dim [0,32) and [32,64): contiguous per lane in global.
  bf16x16 qa0, qa1;
  {
    const int row = qrow_base + (lane & 15);
    const __bf16* qp = &Qb[((size_t)bh * kS + row) * kDH];
    union { uint4 u[2]; bf16x16 v; } c0, c1;
    c0.u[0] = *reinterpret_cast<const uint4*>(qp + hi * 16);
    c0.u[1] = *reinterpret_cast<const uint4*>(qp + 8 + hi * 16);
    c1.u[0] = *reinterpret_cast<const uint4*>(qp + 32 + hi * 16);
    c1.u[1] = *reinterpret_cast<const uint4*>(qp + 40 + hi * 16);
    qa0 = c0.v;
    qa1 = c1.v;
  }
  f32x8 acc[4];
#pragma unroll
  for (int j = 0; j < 4; ++j) acc[j] = zero8();
  float mrow[8], lrow[8];
#pragma unroll
  for (int v = 0; v < 8; ++v) { mrow[v] = -1e30f; lrow[v] = 0.f; }

  const int ktiles = qtile / 32 + 2;  // causal: keys <= qtile+63
  for (int kt = 0; kt < ktiles; ++kt) {
    // stage 32 keys x 64 of K and V, swizzled into fragment order
    {
      const size_t gbase = ((size_t)b * kNKV + kvh) * kS + kt * 32;
      U4 k0v, k1v, v0v, v1v;
      k0v.u = *reinterpret_cast<const uint4*>(&Kb[(gbase + 2 * kp) * kDH + seg * 8]);
      k1v.u = *reinterpret_cast<const uint4*>(&Kb[(gbase + 2 * kp + 1) * kDH + seg * 8]);
      v0v.u = *reinterpret_cast<const uint4*>(&Vb[(gbase + 2 * kp) * kDH + seg * 8]);
      v1v.u = *reinterpret_cast<const uint4*>(&Vb[(gbase + 2 * kp + 1) * kDH + seg * 8]);
      // K: B-fragment for QK^T is (d, key) -> one contiguous b128 per key-row seg
      const int kc = kp >> 3, kpstep = seg >> 2;
      const int khi = ((seg & 3) >> 1) ? 16 : 0, kposi = (seg & 1) * 8;
      const int col0 = (2 * kp) & 15;
      *reinterpret_cast<uint4*>(&Ksw[kc][kpstep][col0 + khi][kposi]) = k0v.u;
      *reinterpret_cast<uint4*>(&Ksw[kc][kpstep][col0 + 1 + khi][kposi]) = k1v.u;
      // V: B-fragment for PV is (key, d) -> pack key pairs into b32 stores
      const int vhi = (kp >= 8) ? 16 : 0, vpos = 2 * (kp & 7);
#pragma unroll
      for (int u = 0; u < 8; ++u) {
        const int d = seg * 8 + u, j = d >> 4, c2 = d & 15;
        *reinterpret_cast<unsigned int*>(&Vsw[j][c2 + vhi][vpos]) =
            (unsigned int)v0v.s[u] | ((unsigned int)v1v.s[u] << 16);
      }
    }
    __syncthreads();

    // scores: 16 rows x 32 keys (two 16-key chunks), K-dim = DH = 64 (2 steps)
    f32x8 s[2];
    s[0] = zero8(); s[1] = zero8();
#pragma unroll
    for (int c = 0; c < 2; ++c) {
      s[c] = wmma_bf16(qa0, load_B_sw(&Ksw[c][0][0][0]), s[c]);
      s[c] = wmma_bf16(qa1, load_B_sw(&Ksw[c][1][0][0]), s[c]);
    }

    const float sscale = 0.125f;  // 1/sqrt(64)
    int am[2];
#pragma unroll
    for (int c = 0; c < 2; ++c) am[c] = amask[b * kS + kt * 32 + c * 16 + col];

    float pv0[8], pv1[8], alpha[8];
#pragma unroll
    for (int v = 0; v < 8; ++v) {
      const int qrow = qrow_base + v + 8 * hi;
      float s0 = s[0][v] * sscale, s1 = s[1][v] * sscale;
      const int k0g = kt * 32 + col, k1g = k0g + 16;
      if (k0g > qrow || am[0] <= 0) s0 = -1e9f;
      if (k1g > qrow || am[1] <= 0) s1 = -1e9f;
      float mx = fmaxf(s0, s1);
#pragma unroll
      for (int msk = 8; msk >= 1; msk >>= 1) mx = fmaxf(mx, __shfl_xor(mx, msk, 16));
      const float mnew = fmaxf(mrow[v], mx);
      const float a = __expf(mrow[v] - mnew);
      const float p0 = __expf(s0 - mnew), p1 = __expf(s1 - mnew);
      float rs = p0 + p1;
#pragma unroll
      for (int msk = 8; msk >= 1; msk >>= 1) rs += __shfl_xor(rs, msk, 16);
      lrow[v] = lrow[v] * a + rs;
      mrow[v] = mnew;
      alpha[v] = a;
      pv0[v] = p0;
      pv1[v] = p1;
    }
#pragma unroll
    for (int j = 0; j < 4; ++j)
#pragma unroll
      for (int v = 0; v < 8; ++v) acc[j][v] *= alpha[v];

    // stage P (C-layout -> A-layout transpose through per-wave LDS slab)
#pragma unroll
    for (int v = 0; v < 8; ++v) {
      Ps[wave][v + 8 * hi][col]      = (__bf16)pv0[v];
      Ps[wave][v + 8 * hi][16 + col] = (__bf16)pv1[v];
    }
    const bf16x16 pa = load_A(&Ps[wave][0][0], 40);
#pragma unroll
    for (int j = 0; j < 4; ++j)
      acc[j] = wmma_bf16(pa, load_B_sw(&Vsw[j][0][0]), acc[j]);
    __syncthreads();
  }

#pragma unroll
  for (int j = 0; j < 4; ++j)
#pragma unroll
    for (int v = 0; v < 8; ++v) {
      const int qrow = qrow_base + v + 8 * hi;
      const float o = acc[j][v] / lrow[v];
      Ob[((size_t)b * kS + qrow) * (kNH * kDH) + h * kDH + j * 16 + col] = (__bf16)o;
    }
}

// ---------------------------------------------------------------------------
// MoE routing
// ---------------------------------------------------------------------------
__global__ __launch_bounds__(256) void route_kernel(
    const __bf16* __restrict__ h2, const float* __restrict__ gw,
    int* __restrict__ counts, int* __restrict__ top_i, float* __restrict__ top_w) {
  const int wave = threadIdx.x >> 5, lane = threadIdx.x & 31;
  const int t = blockIdx.x * 8 + wave;
  float acc[kE];
#pragma unroll
  for (int e = 0; e < kE; ++e) acc[e] = 0.f;
  for (int k = lane; k < kH; k += 32) {
    const float xv = (float)h2[(size_t)t * kH + k];
#pragma unroll
    for (int e = 0; e < kE; ++e) acc[e] += xv * gw[k * kE + e];
  }
#pragma unroll
  for (int e = 0; e < kE; ++e)
#pragma unroll
    for (int m = 16; m >= 1; m >>= 1) acc[e] += __shfl_xor(acc[e], m, 32);
  if (lane == 0) {
    float mx = acc[0];
#pragma unroll
    for (int e = 1; e < kE; ++e) mx = fmaxf(mx, acc[e]);
    float pr[kE], se = 0.f;
#pragma unroll
    for (int e = 0; e < kE; ++e) { pr[e] = __expf(acc[e] - mx); se += pr[e]; }
#pragma unroll
    for (int e = 0; e < kE; ++e) pr[e] /= se;
    int i1 = 0; float w1 = pr[0];
#pragma unroll
    for (int e = 1; e < kE; ++e)
      if (pr[e] > w1) { i1 = e; w1 = pr[e]; }
    int i2 = (i1 == 0) ? 1 : 0; float w2 = pr[i2];
#pragma unroll
    for (int e = 0; e < kE; ++e)
      if (e != i1 && pr[e] > w2) { i2 = e; w2 = pr[e]; }
    top_i[t * 2 + 0] = i1; top_w[t * 2 + 0] = w1;
    top_i[t * 2 + 1] = i2; top_w[t * 2 + 1] = w2;
    atomicAdd(&counts[i1], 1);
    atomicAdd(&counts[i2], 1);
  }
}

__global__ void scan_kernel(const int* __restrict__ counts,
                            int* __restrict__ offsets, int* __restrict__ cursors) {
  if (threadIdx.x == 0 && blockIdx.x == 0) {
    int o = 0;
    for (int e = 0; e < kE; ++e) { offsets[e] = o; cursors[e] = o; o += counts[e]; }
    offsets[kE] = o;
  }
}

__global__ void fill_kernel(const int* __restrict__ top_i,
                            const float* __restrict__ top_w,
                            int* __restrict__ cursors, int* __restrict__ tok_idx,
                            float* __restrict__ tok_wt) {
  const int t = blockIdx.x * 256 + threadIdx.x;
  if (t >= kT) return;
  for (int j = 0; j < 2; ++j) {
    const int e = top_i[t * 2 + j];
    const int pos = atomicAdd(&cursors[e], 1);
    tok_idx[pos] = t;
    tok_wt[pos] = top_w[t * 2 + j];
  }
}

// ---------------------------------------------------------------------------
// MoE up: act[slot, i] = gelu(h2[tok] @ w_in[e]) * (h2[tok] @ w_v[e])
// gathered rows, dual-GEMM in one pass, double-buffered swizzled W tiles.
// 64-row tiles (dual accumulators already use the register budget).
// ---------------------------------------------------------------------------
__global__ __launch_bounds__(256) void moe_up_kernel(
    const __bf16* __restrict__ h2, const float* __restrict__ w_in,
    const float* __restrict__ w_v, const int* __restrict__ offsets,
    const int* __restrict__ tok_idx, __bf16* __restrict__ act) {
  const int e = blockIdx.z;
  const int base = offsets[e];
  const int count = offsets[e + 1] - base;
  const int row0 = blockIdx.y * 64;
  if (row0 >= count) return;
  const int n0 = blockIdx.x * 128;

  __shared__ __attribute__((aligned(16))) __bf16 As[2][64][40];
  __shared__ __attribute__((aligned(16))) __bf16 W1sw[2][8][32][16];
  __shared__ __attribute__((aligned(16))) __bf16 W2sw[2][8][32][16];
  __shared__ int toks[64];

  const int tid = threadIdx.x, lane = tid & 31, wid = tid >> 5;
  if (tid < 64) {
    const int r = row0 + tid;
    toks[tid] = tok_idx[base + (r < count ? r : 0)];
  }
  __syncthreads();

  const float* Wi = w_in + (size_t)e * kH * kI;
  const float* Wv = w_v + (size_t)e * kH * kI;
  const int ar = tid >> 2, ac = (tid & 3) * 8;
  const int kp = tid >> 4, nseg = (tid & 15) * 8;
  const int whi = (kp >= 8) ? 16 : 0, wpos = 2 * (kp & 7);
  const int mt = (wid & 3) * 16, nh = (wid >> 2) * 64, jb = nh >> 4;
  f32x8 acc1[4], acc2[4];
#pragma unroll
  for (int j = 0; j < 4; ++j) { acc1[j] = zero8(); acc2[j] = zero8(); }

  auto stage = [&](int buf, int k0) {
    *reinterpret_cast<uint4*>(&As[buf][ar][ac]) =
        *reinterpret_cast<const uint4*>(&h2[(size_t)toks[ar] * kH + k0 + ac]);
    const size_t wg = (size_t)(k0 + 2 * kp) * kI + n0 + nseg;
    F4 i0, i1, i2, i3, v0, v1, v2, v3;
    i0.v = *reinterpret_cast<const float4*>(&Wi[wg]);
    i1.v = *reinterpret_cast<const float4*>(&Wi[wg + 4]);
    i2.v = *reinterpret_cast<const float4*>(&Wi[wg + kI]);
    i3.v = *reinterpret_cast<const float4*>(&Wi[wg + kI + 4]);
    v0.v = *reinterpret_cast<const float4*>(&Wv[wg]);
    v1.v = *reinterpret_cast<const float4*>(&Wv[wg + 4]);
    v2.v = *reinterpret_cast<const float4*>(&Wv[wg + kI]);
    v3.v = *reinterpret_cast<const float4*>(&Wv[wg + kI + 4]);
#pragma unroll
    for (int u = 0; u < 8; ++u) {
      const int n = nseg + u, j = n >> 4, c = n & 15;
      const float ilo = (u < 4) ? i0.f[u] : i1.f[u - 4];
      const float ihi = (u < 4) ? i2.f[u] : i3.f[u - 4];
      const float vlo = (u < 4) ? v0.f[u] : v1.f[u - 4];
      const float vhi2 = (u < 4) ? v2.f[u] : v3.f[u - 4];
      *reinterpret_cast<unsigned int*>(&W1sw[buf][j][c + whi][wpos]) =
          pack_bf16x2(ilo, ihi);
      *reinterpret_cast<unsigned int*>(&W2sw[buf][j][c + whi][wpos]) =
          pack_bf16x2(vlo, vhi2);
    }
    if (k0 + 32 < kH) {
      __builtin_prefetch(&Wi[(size_t)(k0 + 32 + 2 * kp) * kI + n0 + nseg], 0, 1);
      __builtin_prefetch(&Wv[(size_t)(k0 + 32 + 2 * kp) * kI + n0 + nseg], 0, 1);
    }
  };

  stage(0, 0);
  __syncthreads();
  for (int k0 = 0; k0 < kH; k0 += 32) {
    const int cur = (k0 >> 5) & 1;
    if (k0 + 32 < kH) stage(cur ^ 1, k0 + 32);
    const bf16x16 a = load_A(&As[cur][mt][0], 40);
#pragma unroll
    for (int j = 0; j < 4; ++j) {
      acc1[j] = wmma_bf16(a, load_B_sw(&W1sw[cur][jb + j][0][0]), acc1[j]);
      acc2[j] = wmma_bf16(a, load_B_sw(&W2sw[cur][jb + j][0][0]), acc2[j]);
    }
    __syncthreads();
  }
  const int col = lane & 15, hi = lane >> 4;
#pragma unroll
  for (int j = 0; j < 4; ++j)
#pragma unroll
    for (int v = 0; v < 8; ++v) {
      const int r = mt + v + 8 * hi;
      if (row0 + r < count) {
        const float g = gelu_tanh(acc1[j][v]);
        act[(size_t)(base + row0 + r) * kI + n0 + nh + j * 16 + col] =
            (__bf16)(g * acc2[j][v]);
      }
    }
}

// ---------------------------------------------------------------------------
// MoE down: moe_out[tok] += w_slot * (act[slot] @ w_out[e]); weighted
// scatter-add. 128-row tiles, 32x64 per wave (B fragments feed 2 WMMAs).
// ---------------------------------------------------------------------------
__global__ __launch_bounds__(256) void moe_down_kernel(
    const __bf16* __restrict__ act, const float* __restrict__ w_out,
    const int* __restrict__ offsets, const int* __restrict__ tok_idx,
    const float* __restrict__ tok_wt, float* __restrict__ moe_out) {
  const int e = blockIdx.z;
  const int base = offsets[e];
  const int count = offsets[e + 1] - base;
  const int row0 = blockIdx.y * 128;
  if (row0 >= count) return;
  const int n0 = blockIdx.x * 128;

  __shared__ __attribute__((aligned(16))) __bf16 As[2][128][40];
  __shared__ __attribute__((aligned(16))) __bf16 Wsw[2][8][32][16];
  __shared__ int toks[128];
  __shared__ float wts[128];
  __shared__ int srows[128];

  const int tid = threadIdx.x, lane = tid & 31, wid = tid >> 5;
  if (tid < 128) {
    const int r = row0 + tid;
    const int rr = (r < count ? r : 0);
    toks[tid] = tok_idx[base + rr];
    wts[tid] = tok_wt[base + rr];
    srows[tid] = base + rr;
  }
  __syncthreads();

  const float* Wo = w_out + (size_t)e * kI * kH;
  const int ar = tid >> 1, ac = (tid & 1) * 16;
  const int kp = tid >> 4, nseg = (tid & 15) * 8;
  const int whi = (kp >= 8) ? 16 : 0, wpos = 2 * (kp & 7);
  const int mt = (wid & 3) * 32, nh = (wid >> 2) * 64, jb = nh >> 4;
  f32x8 acc[2][4];
#pragma unroll
  for (int m = 0; m < 2; ++m)
#pragma unroll
    for (int j = 0; j < 4; ++j) acc[m][j] = zero8();

  auto stage = [&](int buf, int k0) {
    const __bf16* asrc = &act[(size_t)srows[ar] * kI + k0 + ac];
    *reinterpret_cast<uint4*>(&As[buf][ar][ac]) =
        *reinterpret_cast<const uint4*>(asrc);
    *reinterpret_cast<uint4*>(&As[buf][ar][ac + 8]) =
        *reinterpret_cast<const uint4*>(asrc + 8);
    const float* Wr0 = &Wo[(size_t)(k0 + 2 * kp) * kH + n0 + nseg];
    const float* Wr1 = Wr0 + kH;
    F4 a0, a1, b0, b1;
    a0.v = *reinterpret_cast<const float4*>(Wr0);
    a1.v = *reinterpret_cast<const float4*>(Wr0 + 4);
    b0.v = *reinterpret_cast<const float4*>(Wr1);
    b1.v = *reinterpret_cast<const float4*>(Wr1 + 4);
#pragma unroll
    for (int u = 0; u < 8; ++u) {
      const float lo = (u < 4) ? a0.f[u] : a1.f[u - 4];
      const float hv = (u < 4) ? b0.f[u] : b1.f[u - 4];
      const int n = nseg + u, j = n >> 4, c = n & 15;
      *reinterpret_cast<unsigned int*>(&Wsw[buf][j][c + whi][wpos]) =
          pack_bf16x2(lo, hv);
    }
    if (k0 + 32 < kI)
      __builtin_prefetch(&Wo[(size_t)(k0 + 32 + 2 * kp) * kH + n0 + nseg], 0, 1);
  };

  stage(0, 0);
  __syncthreads();
  for (int k0 = 0; k0 < kI; k0 += 32) {
    const int cur = (k0 >> 5) & 1;
    if (k0 + 32 < kI) stage(cur ^ 1, k0 + 32);
    const bf16x16 a0 = load_A(&As[cur][mt][0], 40);
    const bf16x16 a1 = load_A(&As[cur][mt + 16][0], 40);
#pragma unroll
    for (int j = 0; j < 4; ++j) {
      const bf16x16 bf = load_B_sw(&Wsw[cur][jb + j][0][0]);
      acc[0][j] = wmma_bf16(a0, bf, acc[0][j]);
      acc[1][j] = wmma_bf16(a1, bf, acc[1][j]);
    }
    __syncthreads();
  }
  const int col = lane & 15, hi = lane >> 4;
#pragma unroll
  for (int m = 0; m < 2; ++m)
#pragma unroll
    for (int j = 0; j < 4; ++j)
#pragma unroll
      for (int v = 0; v < 8; ++v) {
        const int r = mt + m * 16 + v + 8 * hi;
        if (row0 + r < count) {
          atomicAdd(&moe_out[(size_t)toks[r] * kH + n0 + nh + j * 16 + col],
                    wts[r] * acc[m][j][v]);
        }
      }
}

__global__ void zero_kernel(float* __restrict__ p, int n) {
  const int i = blockIdx.x * 256 + threadIdx.x;
  if (i < n) p[i] = 0.f;
}

// ---------------------------------------------------------------------------
// launch
// ---------------------------------------------------------------------------
extern "C" void kernel_launch(void* const* d_in, const int* in_sizes, int n_in,
                              void* d_out, int out_size, void* d_ws, size_t ws_size,
                              hipStream_t stream) {
  (void)in_sizes; (void)n_in; (void)out_size; (void)ws_size;
  const float* hidden   = (const float*)d_in[0];
  const int*   amask    = (const int*)d_in[1];
  const int*   pos_ids  = (const int*)d_in[2];
  const float* q_w      = (const float*)d_in[3];
  const float* k_w      = (const float*)d_in[4];
  const float* v_w      = (const float*)d_in[5];
  const float* o_w      = (const float*)d_in[6];
  const float* pre_attn = (const float*)d_in[7];
  const float* post_attn= (const float*)d_in[8];
  const float* pre_moe  = (const float*)d_in[9];
  const float* post_moe = (const float*)d_in[10];
  const float* gate_w   = (const float*)d_in[11];
  const float* w_in     = (const float*)d_in[12];
  const float* w_v      = (const float*)d_in[13];
  const float* w_out    = (const float*)d_in[14];

  char* ws = (char*)d_ws;
  size_t off = 0;
  auto alloc = [&](size_t bytes) -> void* {
    void* p = ws + off;
    off += (bytes + 255) & ~(size_t)255;
    return p;
  };
  __bf16* h1    = (__bf16*)alloc((size_t)kT * kH * 2);
  float*  qf    = (float*)alloc((size_t)kT * kNH * kDH * 4);
  float*  kf    = (float*)alloc((size_t)kT * kNKV * kDH * 4);
  float*  vf    = (float*)alloc((size_t)kT * kNKV * kDH * 4);
  __bf16* Qb    = (__bf16*)alloc((size_t)kT * kNH * kDH * 2);
  __bf16* Kb    = (__bf16*)alloc((size_t)kT * kNKV * kDH * 2);
  __bf16* Vb    = (__bf16*)alloc((size_t)kT * kNKV * kDH * 2);
  __bf16* attnb = (__bf16*)alloc((size_t)kT * kH * 2);
  float*  attnp = (float*)alloc((size_t)kT * kH * 4);
  float*  xres  = (float*)alloc((size_t)kT * kH * 4);
  __bf16* h2    = (__bf16*)alloc((size_t)kT * kH * 2);
  float*  moe   = (float*)alloc((size_t)kT * kH * 4);
  int*    counts  = (int*)alloc(64 * 4);   // counts[8] | offsets[9] @16 | cursors[8] @32
  int*    offsets = counts + 16;
  int*    cursors = counts + 32;
  int*    topi    = (int*)alloc((size_t)kT * 2 * 4);
  float*  topw    = (float*)alloc((size_t)kT * 2 * 4);
  int*    tok_idx = (int*)alloc((size_t)kT * 2 * 4);
  float*  tok_wt  = (float*)alloc((size_t)kT * 2 * 4);
  __bf16* act     = (__bf16*)alloc((size_t)kT * 2 * kI * 2);

  zero_kernel<<<(kT * kH + 255) / 256, 256, 0, stream>>>(moe, kT * kH);
  zero_kernel<<<1, 64, 0, stream>>>((float*)counts, 64);

  // pre-attn rmsnorm -> bf16
  rms_bf16_kernel<<<kT, 256, 0, stream>>>(hidden, pre_attn, h1);

  // QKV projections (WMMA)
  gemm_bf16_kernel<<<dim3((kNH * kDH) / 128, kT / 128), 256, 0, stream>>>(
      h1, q_w, qf, kNH * kDH, kH);
  gemm_bf16_kernel<<<dim3((kNKV * kDH) / 128, kT / 128), 256, 0, stream>>>(
      h1, k_w, kf, kNKV * kDH, kH);
  gemm_bf16_kernel<<<dim3((kNKV * kDH) / 128, kT / 128), 256, 0, stream>>>(
      h1, v_w, vf, kNKV * kDH, kH);

  // RoPE + pack
  rope_q_kernel<<<(kT * kNH * 32 + 255) / 256, 256, 0, stream>>>(qf, pos_ids, Qb);
  rope_kv_kernel<<<(kT * kNKV * 32 + 255) / 256, 256, 0, stream>>>(kf, vf, pos_ids,
                                                                   Kb, Vb);

  // flash attention
  flash_attn_kernel<<<dim3(kS / 64, kB * kNH), 128, 0, stream>>>(Qb, Kb, Vb, amask,
                                                                 attnb);

  // output projection
  gemm_bf16_kernel<<<dim3(kH / 128, kT / 128), 256, 0, stream>>>(attnb, o_w, attnp,
                                                                 kH, kNH * kDH);

  // post-attn norm + residual + pre-moe norm
  post_attn_kernel<<<kT, 256, 0, stream>>>(attnp, hidden, post_attn, pre_moe, xres, h2);

  // routing
  route_kernel<<<kT / 8, 256, 0, stream>>>(h2, gate_w, counts, topi, topw);
  scan_kernel<<<1, 1, 0, stream>>>(counts, offsets, cursors);
  fill_kernel<<<(kT + 255) / 256, 256, 0, stream>>>(topi, topw, cursors, tok_idx,
                                                    tok_wt);

  // expert GEMMs (gathered)
  moe_up_kernel<<<dim3(kI / 128, kT / 64, kE), 256, 0, stream>>>(h2, w_in, w_v,
                                                                 offsets, tok_idx, act);
  moe_down_kernel<<<dim3(kH / 128, kT / 128, kE), 256, 0, stream>>>(
      act, w_out, offsets, tok_idx, tok_wt, moe);

  // final norm + residual
  final_kernel<<<kT, 256, 0, stream>>>(moe, xres, post_moe, (float*)d_out);
}
